// Attention_2886218023340
// MI455X (gfx1250) — matrix operand order repaired and checked by reading, not compiled, not run on hardware
//
#include <hip/hip_runtime.h>

#define DEVI __device__ __forceinline__

typedef __attribute__((ext_vector_type(16))) __bf16 v16bf;
typedef __attribute__((ext_vector_type(8)))  __bf16 v8bf;
typedef __attribute__((ext_vector_type(8)))  float  v8f;

static constexpr int Bsz   = 8;
static constexpr int Seq   = 1024;
static constexpr int Dim   = 768;
static constexpr int Hn    = 12;
static constexpr int Hd    = 64;
static constexpr int Mrows = Bsz * Seq;          // 8192
static constexpr float kScale = 0.125f;          // 64^-0.5

union AFragU { v16bf v; v8bf h[2]; };

// A-fragment (16x32 bf16, MxK): lane L -> row = row0 + (L%16);
// elements 0..15 -> K = k0 + 16*(L/16) + 0..15
DEVI v16bf load_a_frag(const __bf16* src, int ld, int row0, int k0, int lane) {
    int lo = lane & 15, hi = lane >> 4;
    const __bf16* p = src + (size_t)(row0 + lo) * ld + k0 + hi * 16;
    AFragU a;
    a.h[0] = *(const v8bf*)(p);
    a.h[1] = *(const v8bf*)(p + 8);
    return a.v;
}

// B-fragment (32x16 bf16, KxN) from matrix stored as [N][K] row-major:
// lane L -> col N = n0 + (L%16); elements i -> K = k0 + 16*(L/16) + i
DEVI v16bf load_b_frag(const __bf16* srcT, int ld, int n0, int k0, int lane) {
    int lo = lane & 15, hi = lane >> 4;
    const __bf16* p = srcT + (size_t)(n0 + lo) * ld + k0 + hi * 16;
    return *(const v16bf*)p;
}

DEVI v8f wmma_bf16(v16bf a, v16bf b, v8f c) {
    return __builtin_amdgcn_wmma_f32_16x16x32_bf16(
        /*neg_a=*/false, a, /*neg_b=*/false, b,
        /*c_mod=*/(short)0, c, /*reuse_a=*/false, /*reuse_b=*/false);
}

// ---------------- conversion kernels ----------------

__global__ void cvt_f32_bf16(const float* __restrict__ in,
                             __bf16* __restrict__ out, int n) {
    int i = blockIdx.x * blockDim.x + threadIdx.x;
    if (i < n) out[i] = (__bf16)in[i];
}

// W [Dim x Dim] row-major (k,n)  ->  Wt [n][k] bf16
__global__ void cvt_transpose(const float* __restrict__ W,
                              __bf16* __restrict__ Wt) {
    int i = blockIdx.x * blockDim.x + threadIdx.x;
    if (i < Dim * Dim) {
        int n = i / Dim, k = i % Dim;
        Wt[i] = (__bf16)W[(size_t)k * Dim + n];
    }
}

// ---------------- register-blocked bf16 WMMA GEMM (64x32 per wave) ----------
// C[M,N] = A[M,K] @ Bt[N,K]^T + bias
// out_mode 0: f32 row-major [M, Dim]
// out_mode 1: bf16 head-split  [B, H, S, HD]
// out_mode 2: bf16 head-split-transposed [B, H, HD, S]
static constexpr int MT = 4;   // 16-row tiles per wave  (64 rows)
static constexpr int NT = 2;   // 16-col tiles per wave  (32 cols)

__global__ void __launch_bounds__(32)
gemm_bf16_wmma(const __bf16* __restrict__ A,
               const __bf16* __restrict__ Bt,
               const float* __restrict__ bias,
               float* __restrict__ outF,
               __bf16* __restrict__ outH,
               int K, int out_mode) {
    int lane = threadIdx.x;
    int lo = lane & 15, hi = lane >> 4;
    int n0 = blockIdx.x * (16 * NT);
    int m0 = blockIdx.y * (16 * MT);

    v8f acc[MT][NT] = {};   // 8 independent accumulator chains

    for (int k0 = 0; k0 < K; k0 += 32) {
        v16bf a[MT], b[NT];
#pragma unroll
        for (int i = 0; i < MT; ++i)
            a[i] = load_a_frag(A, K, m0 + 16 * i, k0, lane);
#pragma unroll
        for (int j = 0; j < NT; ++j)
            b[j] = load_b_frag(Bt, K, n0 + 16 * j, k0, lane);
#pragma unroll
        for (int i = 0; i < MT; ++i)
#pragma unroll
            for (int j = 0; j < NT; ++j)
                acc[i][j] = wmma_bf16(a[i], b[j], acc[i][j]);
    }

#pragma unroll
    for (int j = 0; j < NT; ++j) {
        int n = n0 + j * 16 + lo;
        float bval = bias[n];
        int h = n >> 6, hd = n & 63;
#pragma unroll
        for (int i = 0; i < MT; ++i) {
#pragma unroll
            for (int r = 0; r < 8; ++r) {
                int m = m0 + i * 16 + r + 8 * hi;
                float val = acc[i][j][r] + bval;
                if (out_mode == 0) {
                    outF[(size_t)m * Dim + n] = val;
                } else {
                    int bidx = m >> 10, s = m & 1023;
                    size_t addr;
                    if (out_mode == 1)
                        addr = (((size_t)(bidx * Hn + h)) * Seq + s) * Hd + hd;
                    else
                        addr = (((size_t)(bidx * Hn + h)) * Hd + hd) * Seq + s;
                    outH[addr] = (__bf16)val;
                }
            }
        }
    }
}

// ---------------- flash attention: 32 queries (2 q-tiles) per wave ---------
// Qh, Kh : bf16 [B,H,S,HD];  Vt : bf16 [B,H,HD,S];  Obf : bf16 [B*S, Dim]
static constexpr int QT = 2;     // query tiles of 16 per wave

__global__ void __launch_bounds__(32)
attn_wmma(const __bf16* __restrict__ Qh,
          const __bf16* __restrict__ Kh,
          const __bf16* __restrict__ Vt,
          __bf16* __restrict__ Obf) {
    __shared__ __attribute__((aligned(32))) __bf16 Plds[QT][16 * 32];

    int lane = threadIdx.x;
    int lo = lane & 15, hi = lane >> 4;
    int blk = blockIdx.x;
    int qt0 = (blk & 31) * QT;   // S/(16*QT) = 32 wave-tiles along queries
    int bh  = blk >> 5;          // b*H + h

    const __bf16* Q  = Qh + (size_t)bh * Seq * Hd;
    const __bf16* Kp = Kh + (size_t)bh * Seq * Hd;
    const __bf16* Vp = Vt + (size_t)bh * Hd * Seq;

    // Q fragments, head-dim chunks [0,32) and [32,64), kept in registers
    v16bf qa[QT][2];
#pragma unroll
    for (int u = 0; u < QT; ++u) {
        qa[u][0] = load_a_frag(Q, Hd, (qt0 + u) * 16, 0, lane);
        qa[u][1] = load_a_frag(Q, Hd, (qt0 + u) * 16, 32, lane);
    }

    v8f accO[QT][4] = {};          // per q-tile: 16q x 64hd output
    float mrow[QT][8], lrow[QT][8];
#pragma unroll
    for (int u = 0; u < QT; ++u)
#pragma unroll
        for (int r = 0; r < 8; ++r) { mrow[u][r] = -1e30f; lrow[u][r] = 0.f; }

    for (int key0 = 0; key0 < Seq; key0 += 32) {
        // K fragments shared by all q-tiles: keys [key0,key0+32), hd {0,32}
        v16bf kb00 = load_b_frag(Kp, Hd, key0,      0,  lane);
        v16bf kb01 = load_b_frag(Kp, Hd, key0,      32, lane);
        v16bf kb10 = load_b_frag(Kp, Hd, key0 + 16, 0,  lane);
        v16bf kb11 = load_b_frag(Kp, Hd, key0 + 16, 32, lane);

#pragma unroll
        for (int u = 0; u < QT; ++u) {
            v8f s0 = {}, s1 = {};
            s0 = wmma_bf16(qa[u][0], kb00, s0);
            s0 = wmma_bf16(qa[u][1], kb01, s0);
            s1 = wmma_bf16(qa[u][0], kb10, s1);
            s1 = wmma_bf16(qa[u][1], kb11, s1);

            // online softmax; C-frag element r lives in row (r + 8*hi),
            // spread over the 16 lanes of this half-wave.
#pragma unroll
            for (int r = 0; r < 8; ++r) {
                float v0 = s0[r] * kScale;
                float v1 = s1[r] * kScale;
                float rowmax = fmaxf(v0, v1);
#pragma unroll
                for (int w = 1; w < 16; w <<= 1)
                    rowmax = fmaxf(rowmax, __shfl_xor(rowmax, w, 16));
                float newm = fmaxf(mrow[u][r], rowmax);
                float corr = __expf(mrow[u][r] - newm);
                float p0 = __expf(v0 - newm);
                float p1 = __expf(v1 - newm);
                float rowsum = p0 + p1;
#pragma unroll
                for (int w = 1; w < 16; w <<= 1)
                    rowsum += __shfl_xor(rowsum, w, 16);
                lrow[u][r] = lrow[u][r] * corr + rowsum;
                mrow[u][r] = newm;
#pragma unroll
                for (int t = 0; t < 4; ++t) accO[u][t][r] *= corr;
                int row = r + 8 * hi;
                Plds[u][row * 32 + lo]      = (__bf16)p0;   // local key = lo
                Plds[u][row * 32 + 16 + lo] = (__bf16)p1;   // local key = 16+lo
            }
        }
        // cross-lane LDS dependency invisible to the compiler: fence DScnt
        asm volatile("s_wait_dscnt 0" ::: "memory");

        // V fragments shared by all q-tiles; P @ V
        v16bf pa[QT];
#pragma unroll
        for (int u = 0; u < QT; ++u)
            pa[u] = load_a_frag(Plds[u], 32, 0, 0, lane);
#pragma unroll
        for (int t = 0; t < 4; ++t) {
            v16bf vb = load_b_frag(Vp, Seq, t * 16, key0, lane);
#pragma unroll
            for (int u = 0; u < QT; ++u)
                accO[u][t] = wmma_bf16(pa[u], vb, accO[u][t]);
        }
        // LDS reads must retire before next iteration overwrites Plds
        asm volatile("s_wait_dscnt 0" ::: "memory");
    }

    int h = bh % Hn;
    int bidx = bh / Hn;
#pragma unroll
    for (int u = 0; u < QT; ++u) {
#pragma unroll
        for (int t = 0; t < 4; ++t) {
#pragma unroll
            for (int r = 0; r < 8; ++r) {
                int srow = (qt0 + u) * 16 + r + 8 * hi;
                size_t m = (size_t)bidx * Seq + srow;
                int n = h * Hd + t * 16 + lo;
                Obf[m * Dim + n] = (__bf16)(accO[u][t][r] / lrow[u][r]);
            }
        }
    }
}

// ---------------- launcher ----------------

extern "C" void kernel_launch(void* const* d_in, const int* in_sizes, int n_in,
                              void* d_out, int out_size, void* d_ws, size_t ws_size,
                              hipStream_t stream) {
    const float* x  = (const float*)d_in[0];
    const float* Wq = (const float*)d_in[1];
    const float* bq = (const float*)d_in[2];
    const float* Wk = (const float*)d_in[3];
    const float* bk = (const float*)d_in[4];
    const float* Wv = (const float*)d_in[5];
    const float* bv = (const float*)d_in[6];
    const float* Wp = (const float*)d_in[7];
    const float* bp = (const float*)d_in[8];
    float* out = (float*)d_out;

    char* ws = (char*)d_ws;
    size_t off = 0;
    auto alloc = [&](size_t bytes) -> char* {
        char* p = ws + off;
        off += (bytes + 255) & ~(size_t)255;
        return p;
    };

    const size_t xbytes = (size_t)Mrows * Dim * sizeof(__bf16);     // 12.6 MB
    const size_t wbytes = (size_t)Dim * Dim * sizeof(__bf16);       // 1.2 MB

    __bf16* xb  = (__bf16*)alloc(xbytes);
    __bf16* Wqt = (__bf16*)alloc(wbytes);
    __bf16* Wkt = (__bf16*)alloc(wbytes);
    __bf16* Wvt = (__bf16*)alloc(wbytes);
    __bf16* Wpt = (__bf16*)alloc(wbytes);
    __bf16* Qh  = (__bf16*)alloc(xbytes);   // [B,H,S,HD]
    __bf16* Khh = (__bf16*)alloc(xbytes);   // [B,H,S,HD]
    __bf16* Vtt = (__bf16*)alloc(xbytes);   // [B,H,HD,S]
    __bf16* Obf = (__bf16*)alloc(xbytes);   // [B*S, Dim]

    // 1) convert inputs to bf16 (weights pre-transposed to [N][K])
    cvt_f32_bf16<<<(Mrows * Dim + 255) / 256, 256, 0, stream>>>(x, xb, Mrows * Dim);
    cvt_transpose<<<(Dim * Dim + 255) / 256, 256, 0, stream>>>(Wq, Wqt);
    cvt_transpose<<<(Dim * Dim + 255) / 256, 256, 0, stream>>>(Wk, Wkt);
    cvt_transpose<<<(Dim * Dim + 255) / 256, 256, 0, stream>>>(Wv, Wvt);
    cvt_transpose<<<(Dim * Dim + 255) / 256, 256, 0, stream>>>(Wp, Wpt);

    // 2) Q/K/V projections (64x32-per-wave WMMA GEMM), attention layouts
    dim3 gemm_grid(Dim / (16 * NT), Mrows / (16 * MT));   // 24 x 128 wave-tiles
    gemm_bf16_wmma<<<gemm_grid, 32, 0, stream>>>(xb, Wqt, bq, nullptr, Qh,  Dim, 1);
    gemm_bf16_wmma<<<gemm_grid, 32, 0, stream>>>(xb, Wkt, bk, nullptr, Khh, Dim, 1);
    gemm_bf16_wmma<<<gemm_grid, 32, 0, stream>>>(xb, Wvt, bv, nullptr, Vtt, Dim, 2);

    // 3) flash attention: one wave per (b, h, 32-query tile)
    attn_wmma<<<Bsz * Hn * (Seq / (16 * QT)), 32, 0, stream>>>(Qh, Khh, Vtt, Obf);

    // 4) output projection -> f32 result
    gemm_bf16_wmma<<<gemm_grid, 32, 0, stream>>>(Obf, Wpt, bp, out, nullptr, Dim, 0);
}
